// Hybrid_Attention_17669495456352
// MI455X (gfx1250) — compile-verified, matrix operand order checked by reading, not run on hardware
//
#include <hip/hip_runtime.h>
#include <hip/hip_bf16.h>

// ---------------- problem constants ----------------
#define NBK 3
#define BB  4
#define TT  512
#define NN  256
#define DD  1024
#define DXFD 512
#define EE  1024
#define HH  16
#define HDD 64
#define SQ  (NBK*TT)   // 1536
#define SK  (NBK*NN)   // 768
#define EPSF 1e-6f

typedef __attribute__((ext_vector_type(16))) __bf16 v16bf;
typedef __attribute__((ext_vector_type(8)))  float  v8f;
typedef __attribute__((ext_vector_type(4)))  unsigned int v4u_t;
typedef __attribute__((ext_vector_type(8)))  int v8i_t;
typedef __attribute__((ext_vector_type(4)))  int v4i_t;
typedef unsigned short u16;

union FragB { uint4 u4[2]; v16bf v; };

__device__ __forceinline__ u16 f2bf(float f) {
  unsigned int u = __float_as_uint(f);
  unsigned int r = u + 0x7fffu + ((u >> 16) & 1u);   // round-to-nearest-even
  return (u16)(r >> 16);
}

// ---------------- TDM tile DMA: global (strided rows) -> LDS (padded rows) ----------------
// Moves tileH rows x tileW bf16 elements, row stride = strideElems, into LDS with
// hardware padding (padIntervalCode/padAmountCode per D# group1 encoding).
__device__ __forceinline__ void tdm_load_tile(unsigned ldsAddr, const void* gsrc,
                                              unsigned tileW, unsigned tileH,
                                              unsigned strideElems,
                                              unsigned padIntervalCode,
                                              unsigned padAmountCode) {
  unsigned long long ga = (unsigned long long)(size_t)gsrc;
  v4u_t g0;
  g0[0] = 1u;                                               // count=1, user mode
  g0[1] = ldsAddr;                                          // lds_addr
  g0[2] = (unsigned)ga;                                     // global_addr[31:0]
  g0[3] = (unsigned)((ga >> 32) & 0x01ffffffu) | (2u << 30);// global_addr[56:32] | type=2
  const unsigned dim0 = 0x40000000u;                        // huge extents: no OOB clip
  const unsigned dim1 = 0x00100000u;
  int w0 = (1 << 16)                                        // data_size = 2 bytes
         | (1 << 20)                                        // pad_enable
         | ((int)padIntervalCode << 22)
         | ((int)padAmountCode << 25);
  int w1 = (int)((dim0 & 0xffffu) << 16);                   // tensor_dim0[15:0]
  int w2 = (int)(((dim0 >> 16) & 0xffffu) | ((dim1 & 0xffffu) << 16));
  int w3 = (int)(((dim1 >> 16) & 0xffffu) | (tileW << 16)); // tile_dim0
  int w4 = (int)(tileH & 0xffffu);                          // tile_dim1 (tile_dim2=0)
  int w5 = (int)strideElems;                                // tensor_dim0_stride[31:0]
  int w6 = 0, w7 = 0;
  v8i_t g1 = { w0, w1, w2, w3, w4, w5, w6, w7 };
  v4i_t z4 = { 0, 0, 0, 0 };
  v8i_t z8 = { 0, 0, 0, 0, 0, 0, 0, 0 };
  __builtin_amdgcn_tensor_load_to_lds(g0, g1, z4, z4, z8, 0);
}

// ---------------- tiny elementwise kernels ----------------
__global__ __launch_bounds__(256) void silu_kernel(const float* __restrict__ in,
                                                   float* __restrict__ out, int n) {
  int i = blockIdx.x * 256 + threadIdx.x;
  if (i < n) { float v = in[i]; out[i] = v / (1.0f + __expf(-v)); }
}

__global__ __launch_bounds__(256) void cvt_bf16_kernel(const float* __restrict__ in,
                                                       u16* __restrict__ out, int n) {
  int base = (blockIdx.x * 256 + threadIdx.x) * 4;
  #pragma unroll
  for (int j = 0; j < 4; ++j) {
    int i = base + j;
    if (i < n) out[i] = f2bf(in[i]);
  }
}

// ---------------- AdaLN modulation GEMV: out[n,b,j] = silu(emb[b])·W[n,j,:] + bias[n,j]
__global__ __launch_bounds__(128) void mod_gemv(const float* __restrict__ semb,
                                                const float* __restrict__ w,
                                                const float* __restrict__ bias,
                                                float* __restrict__ out,
                                                int D2, int total) {
  int gw = blockIdx.x * 4 + (threadIdx.x >> 5);
  if (gw >= total) return;
  int lane = threadIdx.x & 31;
  int j = gw % D2;
  int nb = gw / D2;
  int b = nb % BB, n = nb / BB;
  const float* wr = w + ((size_t)n * D2 + j) * EE;
  const float* e  = semb + (size_t)b * EE;
  float s = 0.f;
  for (int k = lane; k < EE; k += 32) s += e[k] * wr[k];
  #pragma unroll
  for (int m = 16; m; m >>= 1) s += __shfl_xor(s, m, 32);
  if (lane == 0) out[(size_t)(n * BB + b) * D2 + j] = s + bias[(size_t)n * D2 + j];
}

// ---------------- LayerNorm + AdaLN modulate -> bf16 ----------------
__global__ __launch_bounds__(256) void ln_mod(const float* __restrict__ x,
                                              const float* __restrict__ mod,
                                              u16* __restrict__ out,
                                              int Dm, int L, int nIdx) {
  int row = blockIdx.x;          // b*L + t
  int b = row / L;
  int t = threadIdx.x, lane = t & 31, wv = t >> 5;
  const float* xr = x + (size_t)row * Dm;
  float s = 0.f, s2 = 0.f;
  for (int d = t; d < Dm; d += 256) { float v = xr[d]; s += v; s2 += v * v; }
  #pragma unroll
  for (int m = 16; m; m >>= 1) { s += __shfl_xor(s, m, 32); s2 += __shfl_xor(s2, m, 32); }
  __shared__ float ws1[8], ws2[8];
  if (lane == 0) { ws1[wv] = s; ws2[wv] = s2; }
  __syncthreads();
  float tot = 0.f, tot2 = 0.f;
  #pragma unroll
  for (int i = 0; i < 8; ++i) { tot += ws1[i]; tot2 += ws2[i]; }
  float mu = tot / (float)Dm;
  float var = tot2 / (float)Dm - mu * mu;
  float rstd = rsqrtf(var + EPSF);
  const float* mrow = mod + ((size_t)(nIdx * BB + b)) * (2 * Dm);
  u16* orow = out + (size_t)row * Dm;
  for (int d = t; d < Dm; d += 256) {
    float xn = (xr[d] - mu) * rstd;
    float o = xn * (1.0f + mrow[d]) + mrow[Dm + d];
    orow[d] = f2bf(o);
  }
}

// ---------------- generic WMMA bf16 GEMM, TDM-staged, double-buffered ----------------
// C[row,col] = sum_k A[row,k] * W[col,k] + bias[col]
// store modes: 0 = f32 row-major (ldc); 1 = bf16 head layout Q/K; 2 = bf16 transposed V
__global__ __launch_bounds__(128) void gemm_wmma(const u16* __restrict__ A, int lda, int rpb, long long bsA,
                                                 const u16* __restrict__ W, int ldw,
                                                 const float* __restrict__ bias,
                                                 float* __restrict__ Cf, u16* __restrict__ Cb,
                                                 int Kdim, int mode, int Ls, int Stot, int nIdx, int ldc) {
  __shared__ __align__(16) u16 As[2][64 * 40];   // 64 rows, hw-padded stride 40
  __shared__ __align__(16) u16 Ws[2][64 * 40];
  const int t = threadIdx.x;
  const int lane = t & 31, wave = t >> 5;
  const int wr = wave >> 1, wc = wave & 1;
  const int hi = lane >> 4, nl = lane & 15;
  const int k0 = hi * 8;
  const int rowBase = blockIdx.y * 64;
  const int colBase = blockIdx.x * 64;

  // tile base pointers (64 consecutive rows never cross the batch-segment boundary)
  const u16* Atile = A + (size_t)(rowBase / rpb) * (size_t)bsA + (size_t)(rowBase % rpb) * lda;
  const u16* Wtile = W + (size_t)colBase * ldw;

  unsigned ldsA[2] = { (unsigned)(size_t)&As[0][0], (unsigned)(size_t)&As[1][0] };
  unsigned ldsW[2] = { (unsigned)(size_t)&Ws[0][0], (unsigned)(size_t)&Ws[1][0] };

  v8f acc[2][2];
  #pragma unroll
  for (int i = 0; i < 2; ++i)
    #pragma unroll
    for (int j = 0; j < 2; ++j)
      acc[i][j] = (v8f){0.f,0.f,0.f,0.f,0.f,0.f,0.f,0.f};

  const int nk = Kdim / 32;
  if (t < 32) {   // wave 0 drives the TDM (EXEC ignored by tensor ops)
    tdm_load_tile(ldsA[0], Atile, 32, 64, (unsigned)lda, 3, 3);  // 16-DW rows + 4-DW pad
    tdm_load_tile(ldsW[0], Wtile, 32, 64, (unsigned)ldw, 3, 3);
  }

  for (int ik = 0; ik < nk; ++ik) {
    const int p = ik & 1;
    if (t < 32) __builtin_amdgcn_s_wait_tensorcnt(0);
    __syncthreads();                       // data ready; also drains DS reads of buf p^1
    if (t < 32 && (ik + 1) < nk) {
      tdm_load_tile(ldsA[p ^ 1], Atile + (size_t)(ik + 1) * 32, 32, 64, (unsigned)lda, 3, 3);
      tdm_load_tile(ldsW[p ^ 1], Wtile + (size_t)(ik + 1) * 32, 32, 64, (unsigned)ldw, 3, 3);
    }
    FragB a[2], w[2];
    #pragma unroll
    for (int tr = 0; tr < 2; ++tr) {
      const u16* ptr = As[p] + (wr * 32 + tr * 16 + nl) * 40;
      a[tr].u4[0] = *(const uint4*)(ptr + k0);
      a[tr].u4[1] = *(const uint4*)(ptr + k0 + 16);
    }
    #pragma unroll
    for (int tc = 0; tc < 2; ++tc) {
      const u16* ptr = Ws[p] + (wc * 32 + tc * 16 + nl) * 40;
      w[tc].u4[0] = *(const uint4*)(ptr + k0);
      w[tc].u4[1] = *(const uint4*)(ptr + k0 + 16);
    }
    #pragma unroll
    for (int tr = 0; tr < 2; ++tr)
      #pragma unroll
      for (int tc = 0; tc < 2; ++tc)
        acc[tr][tc] = __builtin_amdgcn_wmma_f32_16x16x32_bf16(
            false, a[tr].v, false, w[tc].v, (short)0, acc[tr][tc], false, false);
  }

  // store
  #pragma unroll
  for (int tr = 0; tr < 2; ++tr) {
    #pragma unroll
    for (int tc = 0; tc < 2; ++tc) {
      int col = colBase + wc * 32 + tc * 16 + nl;
      float bv = bias[col];
      #pragma unroll
      for (int r = 0; r < 8; ++r) {
        int row = rowBase + wr * 32 + tr * 16 + r + hi * 8;
        float val = acc[tr][tc][r] + bv;
        if (mode == 0) {
          Cf[(size_t)row * ldc + col] = val;
        } else {
          int b = row / Ls, s = row % Ls;
          int h = col >> 6, hd = col & 63;
          if (mode == 1)
            Cb[((size_t)(b * HH + h) * Stot + nIdx * Ls + s) * HDD + hd] = f2bf(val);
          else // mode 2: transposed V  Vt[(b,h), hd, s]
            Cb[((size_t)(b * HH + h) * HDD + hd) * Stot + nIdx * Ls + s] = f2bf(val);
        }
      }
    }
  }
}

// ---------------- flash attention (bf16 WMMA, f32 softmax, TDM-staged K/V) ----------------
__global__ __launch_bounds__(128) void attn_kernel(const u16* __restrict__ Qh,
                                                   const u16* __restrict__ Kh,
                                                   const u16* __restrict__ Vt,
                                                   u16* __restrict__ O) {
  __shared__ __align__(16) u16 Ks[2][32 * 72];  // 32 key rows x 64 d, hw-padded stride 72
  __shared__ __align__(16) u16 Vs[2][64 * 40];  // 64 d rows x 32 keys, hw-padded stride 40
  __shared__ __align__(16) u16 Pl[4][16 * 32];  // per-wave P reshape buffer
  const int t = threadIdx.x, lane = t & 31, wave = t >> 5;
  const int hi = lane >> 4, nl = lane & 15;
  const int k0 = hi * 8;
  const int b = blockIdx.z, h = blockIdx.y;
  const int qBase = blockIdx.x * 64 + wave * 16;

  // Q fragments (16 rows x 64 d -> two 16x32 A fragments), held for whole kernel
  const u16* Qrow = Qh + ((size_t)(b * HH + h) * SQ + qBase + nl) * HDD;
  FragB qa[2];
  #pragma unroll
  for (int dh = 0; dh < 2; ++dh) {
    qa[dh].u4[0] = *(const uint4*)(Qrow + dh * 32 + k0);
    qa[dh].u4[1] = *(const uint4*)(Qrow + dh * 32 + k0 + 16);
  }

  v8f o[4];
  #pragma unroll
  for (int i = 0; i < 4; ++i) o[i] = (v8f){0.f,0.f,0.f,0.f,0.f,0.f,0.f,0.f};
  float m_r[8], l_r[8];
  #pragma unroll
  for (int r = 0; r < 8; ++r) { m_r[r] = -3.0e38f; l_r[r] = 0.f; }

  const u16* Kbase = Kh + (size_t)(b * HH + h) * SK * HDD;
  const u16* Vbase = Vt + (size_t)(b * HH + h) * HDD * SK;
  u16* myP = Pl[wave];

  unsigned ldsK[2] = { (unsigned)(size_t)&Ks[0][0], (unsigned)(size_t)&Ks[1][0] };
  unsigned ldsV[2] = { (unsigned)(size_t)&Vs[0][0], (unsigned)(size_t)&Vs[1][0] };

  const int nk = SK / 32;
  if (t < 32) {
    tdm_load_tile(ldsK[0], Kbase, 64, 32, HDD, 4, 3);  // 32-DW rows + 4-DW pad -> stride 72
    tdm_load_tile(ldsV[0], Vbase, 32, 64, SK,  3, 3);  // 16-DW rows + 4-DW pad -> stride 40
  }

  for (int ik = 0; ik < nk; ++ik) {
    const int p = ik & 1;
    const int kc = ik * 32;
    if (t < 32) __builtin_amdgcn_s_wait_tensorcnt(0);
    __syncthreads();
    if (t < 32 && (ik + 1) < nk) {
      tdm_load_tile(ldsK[p ^ 1], Kbase + (size_t)(kc + 32) * HDD, 64, 32, HDD, 4, 3);
      tdm_load_tile(ldsV[p ^ 1], Vbase + (size_t)(kc + 32),       32, 64, SK,  3, 3);
    }

    v8f s0 = (v8f){0.f,0.f,0.f,0.f,0.f,0.f,0.f,0.f};
    v8f s1 = (v8f){0.f,0.f,0.f,0.f,0.f,0.f,0.f,0.f};
    #pragma unroll
    for (int dh = 0; dh < 2; ++dh) {
      FragB kf0, kf1;
      const u16* kr0 = Ks[p] + (size_t)nl * 72 + dh * 32;
      kf0.u4[0] = *(const uint4*)(kr0 + k0);
      kf0.u4[1] = *(const uint4*)(kr0 + k0 + 16);
      const u16* kr1 = Ks[p] + (size_t)(16 + nl) * 72 + dh * 32;
      kf1.u4[0] = *(const uint4*)(kr1 + k0);
      kf1.u4[1] = *(const uint4*)(kr1 + k0 + 16);
      s0 = __builtin_amdgcn_wmma_f32_16x16x32_bf16(false, qa[dh].v, false, kf0.v, (short)0, s0, false, false);
      s1 = __builtin_amdgcn_wmma_f32_16x16x32_bf16(false, qa[dh].v, false, kf1.v, (short)0, s1, false, false);
    }
    // online softmax per accumulator row (rows r / r+8 in opposite lane halves;
    // xor masks < 16 reduce over the 16 columns within each half)
    #pragma unroll
    for (int r = 0; r < 8; ++r) {
      float v0 = s0[r] * 0.125f, v1 = s1[r] * 0.125f;
      float mx = fmaxf(v0, v1);
      #pragma unroll
      for (int mm = 8; mm; mm >>= 1) mx = fmaxf(mx, __shfl_xor(mx, mm, 32));
      float mnew = fmaxf(m_r[r], mx);
      float p0 = __expf(v0 - mnew), p1 = __expf(v1 - mnew);
      float ps = p0 + p1;
      #pragma unroll
      for (int mm = 8; mm; mm >>= 1) ps += __shfl_xor(ps, mm, 32);
      float ratio = __expf(m_r[r] - mnew);
      l_r[r] = l_r[r] * ratio + ps;
      m_r[r] = mnew;
      #pragma unroll
      for (int dt = 0; dt < 4; ++dt) o[dt][r] *= ratio;
      int mrow = r + hi * 8;
      myP[mrow * 32 + nl]      = f2bf(p0);
      myP[mrow * 32 + 16 + nl] = f2bf(p1);
    }
    // reshape P (accumulator layout) -> A-fragment via per-wave LDS (in-order DS, no barrier)
    FragB pf;
    {
      const u16* pr = myP + nl * 32;
      pf.u4[0] = *(const uint4*)(pr + k0);
      pf.u4[1] = *(const uint4*)(pr + k0 + 16);
    }
    #pragma unroll
    for (int dt = 0; dt < 4; ++dt) {
      FragB vf;
      const u16* vr = Vs[p] + (size_t)(dt * 16 + nl) * 40;
      vf.u4[0] = *(const uint4*)(vr + k0);
      vf.u4[1] = *(const uint4*)(vr + k0 + 16);
      o[dt] = __builtin_amdgcn_wmma_f32_16x16x32_bf16(false, pf.v, false, vf.v, (short)0, o[dt], false, false);
    }
  }

  // normalize and store to (B, NB*T, D) bf16
  #pragma unroll
  for (int r = 0; r < 8; ++r) {
    float inv = (l_r[r] > 0.f) ? 1.0f / l_r[r] : 0.f;
    int q = qBase + r + hi * 8;
    #pragma unroll
    for (int dt = 0; dt < 4; ++dt) {
      int col = h * HDD + dt * 16 + nl;
      O[((size_t)b * SQ + q) * DD + col] = f2bf(o[dt][r] * inv);
    }
  }
}

// ---------------- host launcher ----------------
extern "C" void kernel_launch(void* const* d_in, const int* in_sizes, int n_in,
                              void* d_out, int out_size, void* d_ws, size_t ws_size,
                              hipStream_t stream) {
  (void)in_sizes; (void)n_in; (void)out_size; (void)ws_size;
  const float* x[3]  = {(const float*)d_in[0], (const float*)d_in[1], (const float*)d_in[2]};
  const float* xf[3] = {(const float*)d_in[3], (const float*)d_in[4], (const float*)d_in[5]};
  const float* emb   = (const float*)d_in[6];
  const float* adx_w = (const float*)d_in[7];
  const float* adx_b = (const float*)d_in[8];
  const float* adf_w = (const float*)d_in[9];
  const float* adf_b = (const float*)d_in[10];
  const float* qw    = (const float*)d_in[11];
  const float* qb    = (const float*)d_in[12];
  const float* kw    = (const float*)d_in[13];
  const float* kb    = (const float*)d_in[14];
  const float* vw    = (const float*)d_in[15];
  const float* vb    = (const float*)d_in[16];
  const float* ow    = (const float*)d_in[17];
  const float* ob    = (const float*)d_in[18];
  float* out_f = (float*)d_out;

  char* ws = (char*)d_ws;
  size_t off = 0;
  auto take = [&](size_t bytes) -> char* {
    char* p = ws + off;
    off = (off + bytes + 255) & ~(size_t)255;
    return p;
  };
  float* semb    = (float*)take((size_t)BB * EE * 4);
  float* modx    = (float*)take((size_t)NBK * BB * 2 * DD * 4);
  float* modxf   = (float*)take((size_t)NBK * BB * 2 * DXFD * 4);
  u16*   xn_bf   = (u16*)take((size_t)NBK * BB * TT * DD * 2);
  u16*   xfn_bf  = (u16*)take((size_t)NBK * BB * NN * DXFD * 2);
  u16*   qw_bf   = (u16*)take((size_t)NBK * DD * DD * 2);
  u16*   kw_bf   = (u16*)take((size_t)NBK * DD * DXFD * 2);
  u16*   vw_bf   = (u16*)take((size_t)NBK * DD * DXFD * 2);
  u16*   ow_bf   = (u16*)take((size_t)NBK * DD * DD * 2);
  u16*   Qh      = (u16*)take((size_t)BB * HH * SQ * HDD * 2);
  u16*   Kh      = (u16*)take((size_t)BB * HH * SK * HDD * 2);
  u16*   Vt      = (u16*)take((size_t)BB * HH * HDD * SK * 2);
  u16*   attn_bf = (u16*)take((size_t)BB * SQ * DD * 2);

  // 1) silu(emb)
  silu_kernel<<<dim3((BB * EE + 255) / 256), 256, 0, stream>>>(emb, semb, BB * EE);

  // 2) AdaLN modulation GEMVs
  {
    int tot = NBK * BB * 2 * DD;
    mod_gemv<<<dim3((tot + 3) / 4), 128, 0, stream>>>(semb, adx_w, adx_b, modx, 2 * DD, tot);
    int totf = NBK * BB * 2 * DXFD;
    mod_gemv<<<dim3((totf + 3) / 4), 128, 0, stream>>>(semb, adf_w, adf_b, modxf, 2 * DXFD, totf);
  }

  // 3) weight f32 -> bf16
  {
    int n1 = NBK * DD * DD;     // q_w / out_w
    int n2 = NBK * DD * DXFD;   // k_w / v_w
    cvt_bf16_kernel<<<dim3((n1 + 1023) / 1024), 256, 0, stream>>>(qw, qw_bf, n1);
    cvt_bf16_kernel<<<dim3((n2 + 1023) / 1024), 256, 0, stream>>>(kw, kw_bf, n2);
    cvt_bf16_kernel<<<dim3((n2 + 1023) / 1024), 256, 0, stream>>>(vw, vw_bf, n2);
    cvt_bf16_kernel<<<dim3((n1 + 1023) / 1024), 256, 0, stream>>>(ow, ow_bf, n1);
  }

  // 4) LayerNorm + modulate -> bf16
  for (int n = 0; n < NBK; ++n) {
    ln_mod<<<dim3(BB * TT), 256, 0, stream>>>(x[n], modx, xn_bf + (size_t)n * BB * TT * DD, DD, TT, n);
    ln_mod<<<dim3(BB * NN), 256, 0, stream>>>(xf[n], modxf, xfn_bf + (size_t)n * BB * NN * DXFD, DXFD, NN, n);
  }

  // 5) projections (WMMA GEMMs, TDM-staged)
  for (int n = 0; n < NBK; ++n) {
    gemm_wmma<<<dim3(DD / 64, (BB * TT) / 64), 128, 0, stream>>>(
        xn_bf + (size_t)n * BB * TT * DD, DD, BB * TT, (long long)BB * TT * DD,
        qw_bf + (size_t)n * DD * DD, DD, qb + (size_t)n * DD,
        nullptr, Qh, DD, /*mode*/1, /*Ls*/TT, /*Stot*/SQ, n, 0);
    gemm_wmma<<<dim3(DD / 64, (BB * NN) / 64), 128, 0, stream>>>(
        xfn_bf + (size_t)n * BB * NN * DXFD, DXFD, BB * NN, (long long)BB * NN * DXFD,
        kw_bf + (size_t)n * DD * DXFD, DXFD, kb + (size_t)n * DD,
        nullptr, Kh, DXFD, /*mode*/1, /*Ls*/NN, /*Stot*/SK, n, 0);
    gemm_wmma<<<dim3(DD / 64, (BB * NN) / 64), 128, 0, stream>>>(
        xfn_bf + (size_t)n * BB * NN * DXFD, DXFD, BB * NN, (long long)BB * NN * DXFD,
        vw_bf + (size_t)n * DD * DXFD, DXFD, vb + (size_t)n * DD,
        nullptr, Vt, DXFD, /*mode*/2, /*Ls*/NN, /*Stot*/SK, n, 0);
  }

  // 6) attention
  attn_kernel<<<dim3(SQ / 64, HH, BB), 128, 0, stream>>>(Qh, Kh, Vt, attn_bf);

  // 7) output projections -> f32 d_out
  for (int i = 0; i < NBK; ++i) {
    gemm_wmma<<<dim3(DD / 64, (BB * TT) / 64), 128, 0, stream>>>(
        attn_bf + (size_t)i * TT * DD, DD, /*rpb*/TT, /*bsA*/(long long)SQ * DD,
        ow_bf + (size_t)i * DD * DD, DD, ob + (size_t)i * DD,
        out_f + (size_t)i * BB * TT * DD, nullptr, DD, /*mode*/0, TT, SQ, i, DD);
  }
}
